// MOELayer_38044820308414
// MI455X (gfx1250) — compile-verified, matrix operand order checked by reading, not run on hardware
//
#include <hip/hip_runtime.h>
#include <hip/hip_bf16.h>

// ---------------- problem constants (from reference shapes) ----------------
#define S_TOK 32768
#define MDIM  512
#define NEXP  64
#define VDIM  2048
#define TOPK  2
#define CAP   1024   // TOP_K * ceil(S/E) = 2 * 512

// GEMM tiling
#define BROWS 64     // block rows (over CAP)
#define BCOLS 256    // block cols (over output dim)
#define KSLAB 32     // K per stage (one WMMA K)
#define LPAD  40     // padded LDS row stride in elements (80B = 20 banks, conflict-free)

// ---------------- WMMA fragment types ----------------
typedef __attribute__((ext_vector_type(16))) __bf16       v16bf;
typedef __attribute__((ext_vector_type(8)))  float        v8f;
typedef __attribute__((ext_vector_type(4)))  unsigned int u32x4;
typedef __attribute__((ext_vector_type(4)))  int          v4i;

// address-space-qualified pointer types for the async builtin
typedef __attribute__((address_space(1))) v4i* as1_v4i_ptr;
typedef __attribute__((address_space(3))) v4i* as3_v4i_ptr;

union Frag {
    v16bf v;
    u32x4 u[2];
};

// per-lane fragment: elements 0..7 = K(ko..ko+7), elements 8..15 = K(ko+16..ko+23)
__device__ __forceinline__ void load_frag(Frag& f, const __bf16* p) {
    f.u[0] = *reinterpret_cast<const u32x4*>(p);
    f.u[1] = *reinterpret_cast<const u32x4*>(p + 16);
}

// ---------------- gfx1250 async global->LDS copy (ASYNCcnt path) ----------------
#if __has_builtin(__builtin_amdgcn_global_load_async_to_lds_b128)
#define USE_ASYNC_LDS 1
#else
#define USE_ASYNC_LDS 0
#endif

__device__ __forceinline__ void cp_async16(__bf16* l, const __bf16* g) {
#if USE_ASYNC_LDS
    void* gv = (void*)g; // drop const; generic -> AS(1)/AS(3) casts below
    void* lv = (void*)l;
    __builtin_amdgcn_global_load_async_to_lds_b128((as1_v4i_ptr)gv, (as3_v4i_ptr)lv, 0, 0);
#else
    *reinterpret_cast<u32x4*>(l) = *reinterpret_cast<const u32x4*>(g);
#endif
}

__device__ __forceinline__ void wait_async_group() { // allow the 5 just-issued to fly
#if USE_ASYNC_LDS
#if __has_builtin(__builtin_amdgcn_s_wait_asynccnt)
    __builtin_amdgcn_s_wait_asynccnt(5);
#else
    asm volatile("s_wait_asynccnt 0x5" ::: "memory");
#endif
#endif
}

__device__ __forceinline__ void wait_async_all() {
#if USE_ASYNC_LDS
#if __has_builtin(__builtin_amdgcn_s_wait_asynccnt)
    __builtin_amdgcn_s_wait_asynccnt(0);
#else
    asm volatile("s_wait_asynccnt 0x0" ::: "memory");
#endif
#endif
}

// ---------------- utility kernels ----------------
__global__ void zero_u32_kernel(unsigned int* __restrict__ p, size_t n) {
    size_t i = (size_t)blockIdx.x * blockDim.x + threadIdx.x;
    if (i < n) p[i] = 0u;
}

// fc1_w [E][M][V] f32  ->  fc1t [E][V][M] bf16  (B pre-transposed: K=M contiguous)
__global__ void conv_fc1_kernel(const float* __restrict__ w, __bf16* __restrict__ wt) {
    size_t idx = (size_t)blockIdx.x * blockDim.x + threadIdx.x; // < E*V*M
    int    m   = (int)(idx % MDIM);
    size_t rem = idx / MDIM;
    int    v   = (int)(rem % VDIM);
    int    e   = (int)(rem / VDIM);
    wt[idx] = (__bf16)w[((size_t)e * MDIM + m) * VDIM + v];
}

// fc2_w [E][V][M] f32  ->  fc2t [E][M][V] bf16  (K=V contiguous)
__global__ void conv_fc2_kernel(const float* __restrict__ w, __bf16* __restrict__ wt) {
    size_t idx = (size_t)blockIdx.x * blockDim.x + threadIdx.x; // < E*M*V
    int    v   = (int)(idx % VDIM);
    size_t rem = idx / VDIM;
    int    m   = (int)(rem % MDIM);
    int    e   = (int)(rem / MDIM);
    wt[idx] = (__bf16)w[((size_t)e * VDIM + v) * MDIM + m];
}

// ---------------- gating: logits, softmax, top-2, aux-loss stats ----------------
__global__ void gate_kernel(const float* __restrict__ x, const float* __restrict__ wg,
                            int* __restrict__ topk, float* __restrict__ gates_n,
                            float* __restrict__ me, float* __restrict__ ce) {
    __shared__ float xs[MDIM];
    __shared__ float lg[NEXP];
    const int s = blockIdx.x;
    const int t = threadIdx.x; // 0..63, expert id
    for (int m = t; m < MDIM; m += 64) xs[m] = x[(size_t)s * MDIM + m];
    __syncthreads();

    const float* w = wg + (size_t)t * MDIM;
    float acc = 0.f;
#pragma unroll 8
    for (int m = 0; m < MDIM; ++m) acc += xs[m] * w[m];
    lg[t] = acc;
    __syncthreads();

    float maxv = -INFINITY;
    for (int e = 0; e < NEXP; ++e) maxv = fmaxf(maxv, lg[e]);
    float sum = 0.f;
    for (int e = 0; e < NEXP; ++e) sum += __expf(lg[e] - maxv);

    atomicAdd(&me[t], __expf(lg[t] - maxv) / sum);

    if (t == 0) {
        float v1 = -INFINITY, v2 = -INFINITY;
        int   i1 = 0, i2 = 0;
        for (int e = 0; e < NEXP; ++e) {
            float l = lg[e];
            if (l > v1)      { v2 = v1; i2 = i1; v1 = l; i1 = e; }
            else if (l > v2) { v2 = l; i2 = e; }
        }
        float g1  = __expf(v1 - maxv) / sum;
        float g2  = __expf(v2 - maxv) / sum;
        float den = fmaxf(g1 + g2, 1.1920929e-07f);
        topk[s * 2]        = i1;
        topk[s * 2 + 1]    = i2;
        gates_n[s * 2]     = g1 / den;
        gates_n[s * 2 + 1] = g2 / den;
        atomicAdd(&ce[i1], 1.0f); // masks[:,0,:] count
    }
}

// ---------------- capacity locations: ordered ballot scan, 1 wave per expert ----------------
__global__ void locs_kernel(const int* __restrict__ topk, int* __restrict__ locs) {
    const int e    = blockIdx.x;
    const int lane = threadIdx.x;               // wave32
    const unsigned lanemask = (1u << lane) - 1u;
    int base = 0;                               // k=1 continues after k=0 (prev_counts)
    for (int k = 0; k < TOPK; ++k) {
        for (int s0 = 0; s0 < S_TOK; s0 += 32) {
            const int  s = s0 + lane;
            const bool f = (topk[s * TOPK + k] == e);
            const unsigned m = (unsigned)__ballot(f);
            if (f) locs[s * TOPK + k] = base + __popc(m & lanemask);
            base += __popc(m);
        }
    }
}

// ---------------- scatter tokens into disp[E][CAP][M] (bf16) ----------------
__global__ void scatter_kernel(const float* __restrict__ x, const int* __restrict__ topk,
                               const int* __restrict__ locs, __bf16* __restrict__ disp) {
    size_t idx = (size_t)blockIdx.x * blockDim.x + threadIdx.x; // < S*K*M
    int m  = (int)(idx & (MDIM - 1));
    int sk = (int)(idx / MDIM);
    int loc = locs[sk];
    if (loc < CAP) {
        int e = topk[sk];
        int s = sk >> 1;
        disp[((size_t)e * CAP + loc) * MDIM + m] = (__bf16)x[(size_t)s * MDIM + m];
    }
}

// ---------------- WMMA GEMM with async-LDS double-buffered staging ----------------
// A: [E][CAP][KD] bf16 row-major (K contiguous)
// B: [E][OUTLD][KD] bf16 pre-transposed (N rows, K contiguous)
// O: [E][CAP][OUTLD]
// Block: 256 threads (8 waves, 2 row-groups x 4 col-groups). Block tile 64 x 256.
// Per K-slab (32): DMA A-slab (64x32) + B-slab (256x32) into LDS (padded rows),
// double-buffered; waves read fragments via ds_load_b128 and issue 8 WMMAs each.
template <int KD, int OUTLD, bool RELU, typename OutT>
__global__ void __launch_bounds__(256) gemm_kernel(const __bf16* __restrict__ A,
                                                   const __bf16* __restrict__ B,
                                                   const float*  __restrict__ bias,
                                                   OutT* __restrict__ O) {
    __shared__ __align__(16) __bf16 sA[2][BROWS * LPAD];  // 2 x 5 KB
    __shared__ __align__(16) __bf16 sB[2][BCOLS * LPAD];  // 2 x 20 KB

    const int e    = blockIdx.z;
    const int t    = threadIdx.x;
    const int wave = t >> 5;
    const int lane = t & 31;
    const int rowBlock = blockIdx.y * BROWS;
    const int colBlock = blockIdx.x * BCOLS;
    const int hl = lane & 15;          // A-row / B-col within 16-tile
    const int ko = (lane >> 4) << 3;   // K sub-offset 0 or 8

    // cooperative-copy coordinates: 256 threads = 64 rows x 4 chunks of 16B
    const int trow = t >> 2;           // 0..63
    const int tchk = (t & 3) * 8;      // element offset 0,8,16,24

    const __bf16* gA = A + (size_t)e * CAP * KD   + (size_t)(rowBlock + trow) * KD + tchk;
    const __bf16* gB = B + (size_t)e * OUTLD * KD + (size_t)(colBlock + trow) * KD + tchk;

    auto stage = [&](int buf, int k0) { // 5 async wave-instructions per wave
        cp_async16(&sA[buf][trow * LPAD + tchk], gA + k0);
#pragma unroll
        for (int p = 0; p < 4; ++p)
            cp_async16(&sB[buf][(p * 64 + trow) * LPAD + tchk],
                       gB + (size_t)(p * 64) * KD + k0);
    };

    v8f acc[2][4] = {};
    const int arow0 = (wave & 1) * 32;   // wave's 32 rows within block tile
    const int bcol0 = (wave >> 1) * 64;  // wave's 64 cols within block tile

    auto mma_slab = [&](int buf) {
        Frag a[2], b[4];
#pragma unroll
        for (int i = 0; i < 2; ++i)
            load_frag(a[i], &sA[buf][(arow0 + i * 16 + hl) * LPAD + ko]);
#pragma unroll
        for (int j = 0; j < 4; ++j)
            load_frag(b[j], &sB[buf][(bcol0 + j * 16 + hl) * LPAD + ko]);
#pragma unroll
        for (int i = 0; i < 2; ++i)
#pragma unroll
            for (int j = 0; j < 4; ++j)
                acc[i][j] = __builtin_amdgcn_wmma_f32_16x16x32_bf16(
                    false, a[i].v, false, b[j].v, (short)0, acc[i][j], false, false);
    };

    stage(0, 0);
    int buf = 0;
    for (int k0 = 0; k0 < KD; k0 += KSLAB) {
        const int nk = k0 + KSLAB;
        if (nk < KD) {
            if (nk + 128 < KD) { // pull the streams ~4 slabs ahead into cache
                __builtin_prefetch(gA + nk + 128, 0, 3);
                __builtin_prefetch(gB + nk + 128, 0, 3);
            }
            stage(buf ^ 1, nk);      // DMA next slab while current computes
            wait_async_group();      // own slab-k copies done; next 5 stay in flight
        } else {
            wait_async_all();
        }
        __syncthreads();             // all waves' slab-k copies visible
        mma_slab(buf);
        __syncthreads();             // all reads done before buffer is re-staged
        buf ^= 1;
    }

    // epilogue: D layout -> row = base + r (+16 per i), col = base + 16*j
    const int orow = rowBlock + arow0 + ((lane >> 4) << 3);
    const int ocol = colBlock + bcol0 + hl;
    const size_t obase = (size_t)e * CAP * OUTLD;
#pragma unroll
    for (int i = 0; i < 2; ++i) {
#pragma unroll
        for (int j = 0; j < 4; ++j) {
            const int   c  = ocol + j * 16;
            const float bv = bias[(size_t)e * OUTLD + c];
#pragma unroll
            for (int r = 0; r < 8; ++r) {
                float v = acc[i][j][r] + bv;
                if (RELU) v = fmaxf(v, 0.f);
                O[obase + (size_t)(orow + i * 16 + r) * OUTLD + c] = (OutT)v;
            }
        }
    }
}

// ---------------- combine/gather ----------------
__global__ void combine_kernel(const float* __restrict__ eout, const int* __restrict__ topk,
                               const int* __restrict__ locs, const float* __restrict__ gates_n,
                               float* __restrict__ y) {
    size_t idx = (size_t)blockIdx.x * blockDim.x + threadIdx.x; // < S*M
    int m = (int)(idx & (MDIM - 1));
    int s = (int)(idx / MDIM);
    float acc = 0.f;
#pragma unroll
    for (int k = 0; k < TOPK; ++k) {
        int loc = locs[s * 2 + k];
        if (loc < CAP) {
            int e = topk[s * 2 + k];
            acc += gates_n[s * 2 + k] * eout[((size_t)e * CAP + loc) * MDIM + m];
        }
    }
    y[idx] = acc;
}

// ---------------- aux loss scalar ----------------
__global__ void loss_kernel(const float* __restrict__ me, const float* __restrict__ ce,
                            float* __restrict__ out) {
    __shared__ float r[NEXP];
    int t = threadIdx.x;
    r[t] = me[t] * ce[t];
    __syncthreads();
    if (t == 0) {
        float s = 0.f;
        for (int i = 0; i < NEXP; ++i) s += r[i];
        out[(size_t)S_TOK * MDIM] = s * ((float)NEXP / ((float)S_TOK * (float)S_TOK));
    }
}

// ---------------- host launch ----------------
extern "C" void kernel_launch(void* const* d_in, const int* in_sizes, int n_in,
                              void* d_out, int out_size, void* d_ws, size_t ws_size,
                              hipStream_t stream) {
    (void)in_sizes; (void)n_in; (void)out_size; (void)ws_size;
    const float* x     = (const float*)d_in[0]; // [S, M]
    const float* wg    = (const float*)d_in[1]; // [E, M]
    const float* fc1_w = (const float*)d_in[2]; // [E, M, V]
    const float* fc1_b = (const float*)d_in[3]; // [E, V]
    const float* fc2_w = (const float*)d_in[4]; // [E, V, M]
    const float* fc2_b = (const float*)d_in[5]; // [E, M]
    float* y = (float*)d_out;                   // [S*M] + 1 (loss)

    // workspace carve-up (256B aligned)
    char* ws = (char*)d_ws;
    size_t off = 0;
    auto take = [&](size_t bytes) -> void* {
        void* p = ws + off;
        off = (off + bytes + 255) & ~(size_t)255;
        return p;
    };
    int*     topk    = (int*)   take((size_t)S_TOK * TOPK * sizeof(int));
    int*     locs    = (int*)   take((size_t)S_TOK * TOPK * sizeof(int));
    float*   gates_n = (float*) take((size_t)S_TOK * TOPK * sizeof(float));
    float*   mece    = (float*) take(2 * NEXP * sizeof(float)); // me[64] | ce[64]
    __bf16*  disp    = (__bf16*)take((size_t)NEXP * CAP * MDIM * sizeof(__bf16));
    __bf16*  fc1t    = (__bf16*)take((size_t)NEXP * VDIM * MDIM * sizeof(__bf16));
    __bf16*  fc2t    = (__bf16*)take((size_t)NEXP * MDIM * VDIM * sizeof(__bf16));
    __bf16*  h       = (__bf16*)take((size_t)NEXP * CAP * VDIM * sizeof(__bf16));
    float*   eout    = (float*) take((size_t)NEXP * CAP * MDIM * sizeof(float));
    float* me = mece;
    float* ce = mece + NEXP;

    const int T = 256;

    // 1) zero me/ce and disp
    zero_u32_kernel<<<1, T, 0, stream>>>((unsigned int*)mece, 2 * NEXP);
    {
        size_t nw = (size_t)NEXP * CAP * MDIM * sizeof(__bf16) / 4;
        zero_u32_kernel<<<(unsigned)((nw + T - 1) / T), T, 0, stream>>>((unsigned int*)disp, nw);
    }

    // 2) transpose-convert weights to bf16
    {
        size_t n = (size_t)NEXP * MDIM * VDIM;
        conv_fc1_kernel<<<(unsigned)((n + T - 1) / T), T, 0, stream>>>(fc1_w, fc1t);
        conv_fc2_kernel<<<(unsigned)((n + T - 1) / T), T, 0, stream>>>(fc2_w, fc2t);
    }

    // 3) gating
    gate_kernel<<<S_TOK, NEXP, 0, stream>>>(x, wg, topk, gates_n, me, ce);

    // 4) capacity locations (ordered scan, 1 wave32 per expert)
    locs_kernel<<<NEXP, 32, 0, stream>>>(topk, locs);

    // 5) scatter tokens -> disp (bf16)
    {
        size_t n = (size_t)S_TOK * TOPK * MDIM;
        scatter_kernel<<<(unsigned)((n + T - 1) / T), T, 0, stream>>>(x, topk, locs, disp);
    }

    // 6) expert FFN via WMMA bf16 + async-LDS staging
    gemm_kernel<MDIM, VDIM, true, __bf16>
        <<<dim3(VDIM / BCOLS, CAP / BROWS, NEXP), T, 0, stream>>>(disp, fc1t, fc1_b, h);
    gemm_kernel<VDIM, MDIM, false, float>
        <<<dim3(MDIM / BCOLS, CAP / BROWS, NEXP), T, 0, stream>>>(h, fc2t, fc2_b, eout);

    // 7) combine + loss
    {
        size_t n = (size_t)S_TOK * MDIM;
        combine_kernel<<<(unsigned)((n + T - 1) / T), T, 0, stream>>>(eout, topk, locs, gates_n, y);
    }
    loss_kernel<<<1, NEXP, 0, stream>>>(me, ce, y);
}
